// LSTM_68959994904825
// MI455X (gfx1250) — matrix operand order repaired and checked by reading, not compiled, not run on hardware
//
#include <hip/hip_runtime.h>
#include <hip/hip_bf16.h>

// ---------------------------------------------------------------------------
// LSTM (B=128, S=1024, IN=512, H=512) + Dense(1) for MI455X (gfx1250).
// Persistent cooperative kernel: bf16 WMMA GEMMs, weights packed into WMMA
// B-layout and parked in LDS for all 1024 steps, one grid barrier per step.
// x is pre-converted to bf16; h is kept in bf16 so the recurrent loop does
// zero dtype conversion work (pure load -> v_wmma).
// ---------------------------------------------------------------------------

typedef __attribute__((ext_vector_type(16))) __bf16 v16bf;
typedef __attribute__((ext_vector_type(8)))  __bf16 v8bf;
typedef __attribute__((ext_vector_type(8)))  float  v8f;
typedef __attribute__((ext_vector_type(4)))  float  f4;

#define BB    128
#define SEQ   1024
#define INF   512
#define HID   512
#define G4    2048
#define NBLK  64

#define WP_ELEMS   (128 * 32 * 32 * 16)        // packed bf16 weight elements (4 MB)
#define X_ELEMS    (BB * SEQ * INF)            // 67,108,864 -> 128 MB bf16
#define LDS_B_BYTES (4 * 32 * 32 * 16 * 2)     // 128 KB packed weights
#define LDS_G_BYTES (4 * 64 * 16 * 4)          // 16 KB gate staging

// ---------------------------------------------------------------------------
// Pack Wi (512x2048) and Wh (512x2048) fp32 row-major into bf16 WMMA B-tile
// layout: [ntile(128)][ktile(32: 0-15=Wi,16-31=Wh)][lane(32)][j(16)].
// ISA 16-bit B 32x16 layout: col=lane&15, h=lane>>4; j<8 -> K=8h+j,
// j>=8 -> K=16+8h+(j-8).  Each lane's 16 bf16 are contiguous (32B).
// ---------------------------------------------------------------------------
__global__ void pack_weights_kernel(const float* __restrict__ Wi,
                                    const float* __restrict__ Wh,
                                    __bf16* __restrict__ Wp) {
    int idx = blockIdx.x * blockDim.x + threadIdx.x;
    if (idx >= WP_ELEMS) return;
    int j    = idx & 15;
    int lane = (idx >> 4) & 31;
    int kt   = (idx >> 9) & 31;
    int nt   = idx >> 14;
    int col  = nt * 16 + (lane & 15);
    int krow = ((lane >> 4) << 3) + (j & 7) + ((j >> 3) << 4);  // 8h+(j&7)+(j>=8?16:0)
    float v;
    if (kt < 16) v = Wi[(size_t)(kt * 32 + krow) * G4 + col];
    else         v = Wh[(size_t)((kt - 16) * 32 + krow) * G4 + col];
    Wp[idx] = (__bf16)v;
}

// x fp32 -> bf16, same row-major layout. 4 elements per thread.
__global__ void convert_x_kernel(const float* __restrict__ x,
                                 __bf16* __restrict__ xb) {
    size_t i = ((size_t)blockIdx.x * blockDim.x + threadIdx.x) * 4;
    f4 v = *(const f4*)(x + i);
    v8bf* dst = (v8bf*)0;  (void)dst;
    __bf16 o0 = (__bf16)v.x, o1 = (__bf16)v.y, o2 = (__bf16)v.z, o3 = (__bf16)v.w;
    xb[i + 0] = o0; xb[i + 1] = o1; xb[i + 2] = o2; xb[i + 3] = o3;
}

__global__ void init_state_kernel(unsigned* __restrict__ h32,   // hA||hB as u32
                                  float* __restrict__ c,
                                  int* __restrict__ bar) {
    int i = blockIdx.x * blockDim.x + threadIdx.x;
    if (i < BB * HID) c[i] = 0.f;
    if (i < BB * HID)  h32[i] = 0u;   // covers both bf16 h buffers (2*64K*2B)
    if (i < 8) bar[i] = 0;
}

__device__ __forceinline__ float sigm(float v) {
    return 1.0f / (1.0f + __expf(-v));
}
__device__ __forceinline__ float fast_tanh(float v) {
    float e = __expf(2.0f * v);
    return (e - 1.0f) / (e + 1.0f);
}

// ---------------------------------------------------------------------------
// Persistent scan kernel. Grid = 64 blocks x 256 threads.
// block = (ng = blockIdx.x & 31  -> gate n-group {ng, ng+32, ng+64, ng+96},
//          mh = blockIdx.x >> 5  -> batch rows [mh*64, mh*64+64)).
// wave w: M-tile = mh*4 + (w>>1); gates = {(w&1)*2, (w&1)*2+1}.
// ---------------------------------------------------------------------------
__launch_bounds__(256, 1)
__global__ void lstm_persistent_kernel(const __bf16* __restrict__ xb,
                                       const float* __restrict__ bias,
                                       const float* __restrict__ Wd,
                                       const float* __restrict__ bd,
                                       const __bf16* __restrict__ Wp,
                                       __bf16* __restrict__ hA,
                                       __bf16* __restrict__ hB,
                                       float* __restrict__ c,
                                       float* __restrict__ out,
                                       int* __restrict__ bar) {
    extern __shared__ char smem[];
    __bf16* Bs = (__bf16*)smem;                                      // 128 KB
    float (*gbuf)[64][16] = (float (*)[64][16])(smem + LDS_B_BYTES); // 16 KB

    const int tid   = threadIdx.x;
    const int lane  = tid & 31;
    const int wave  = tid >> 5;
    const int ng    = blockIdx.x & 31;
    const int mh    = blockIdx.x >> 5;
    const int mtl   = wave >> 1;          // local M-tile 0..3
    const int gpair = (wave & 1) * 2;     // first gate of this wave's pair

    // ---- stage packed weights (4 n-tiles x 32 k-tiles) into LDS, once ----
    for (int g = 0; g < 4; ++g) {
        int nt = ng + 32 * g;
        const uint4* src = (const uint4*)(Wp + (size_t)nt * (32 * 32 * 16));
        uint4* dst = (uint4*)(Bs + g * (32 * 32 * 16));
        for (int i = tid; i < (32 * 32 * 16 * 2) / 16; i += 256) dst[i] = src[i];
    }
    __syncthreads();

    // ---- per-lane constants ----
    const int mrow = mh * 64 + mtl * 16 + (lane & 15);  // batch row this lane loads
    const int ksel = (lane >> 4) * 8;                   // A-layout K select (0 / 8)
    const __bf16* xbase = xb + (size_t)mrow * SEQ * INF;
    const float bias0 = bias[(ng + 32 * (gpair + 0)) * 16 + (lane & 15)];
    const float bias1 = bias[(ng + 32 * (gpair + 1)) * 16 + (lane & 15)];

    for (int t = 0; t < SEQ; ++t) {
        const __bf16* __restrict__ hr = (t & 1) ? hB : hA;  // read h_{t-1}
        __bf16* __restrict__ hw       = (t & 1) ? hA : hB;  // write h_t

        v8f acc0, acc1;
        #pragma unroll
        for (int r = 0; r < 8; ++r) { acc0[r] = bias0; acc1[r] = bias1; }

        const __bf16* xrow = xbase + (size_t)t * INF;
        const __bf16* hrow = hr + (size_t)mrow * HID;

        // ---- x_t @ Wi : k-tiles 0..15 ----
        #pragma unroll 8
        for (int kt = 0; kt < 16; ++kt) {
            const int k0 = (kt << 5) + ksel;
            v8bf c0 = *(const v8bf*)(xrow + k0);        // K [k0, k0+8)
            v8bf c1 = *(const v8bf*)(xrow + k0 + 16);   // K [k0+16, k0+24)
            v16bf a = __builtin_shufflevector(c0, c1,
                0, 1, 2, 3, 4, 5, 6, 7, 8, 9, 10, 11, 12, 13, 14, 15);
            const v16bf b0 = *(const v16bf*)(Bs + (((gpair + 0) * 32 + kt) * 32 + lane) * 16);
            const v16bf b1 = *(const v16bf*)(Bs + (((gpair + 1) * 32 + kt) * 32 + lane) * 16);
            acc0 = __builtin_amdgcn_wmma_f32_16x16x32_bf16(
                false, a, false, b0, (short)0, acc0, false, false);
            acc1 = __builtin_amdgcn_wmma_f32_16x16x32_bf16(
                false, a, false, b1, (short)0, acc1, false, false);
        }

        // ---- h_{t-1} @ Wh : k-tiles 16..31 ----
        #pragma unroll 8
        for (int kt = 0; kt < 16; ++kt) {
            const int k0 = (kt << 5) + ksel;
            v8bf c0 = *(const v8bf*)(hrow + k0);
            v8bf c1 = *(const v8bf*)(hrow + k0 + 16);
            v16bf a = __builtin_shufflevector(c0, c1,
                0, 1, 2, 3, 4, 5, 6, 7, 8, 9, 10, 11, 12, 13, 14, 15);
            const v16bf b0 = *(const v16bf*)(Bs + (((gpair + 0) * 32 + 16 + kt) * 32 + lane) * 16);
            const v16bf b1 = *(const v16bf*)(Bs + (((gpair + 1) * 32 + 16 + kt) * 32 + lane) * 16);
            acc0 = __builtin_amdgcn_wmma_f32_16x16x32_bf16(
                false, a, false, b0, (short)0, acc0, false, false);
            acc1 = __builtin_amdgcn_wmma_f32_16x16x32_bf16(
                false, a, false, b1, (short)0, acc1, false, false);
        }

        // ---- dump gate accumulators to LDS (C layout: VGPR r -> M=r(+8)) ----
        {
            const int mbase = mtl * 16 + ((lane < 16) ? 0 : 8);
            const int n = lane & 15;
            #pragma unroll
            for (int r = 0; r < 8; ++r) {
                gbuf[gpair + 0][mbase + r][n] = acc0[r];
                gbuf[gpair + 1][mbase + r][n] = acc1[r];
            }
        }
        __syncthreads();

        // ---- elementwise LSTM cell update: 64 rows x 16 cols per block ----
        for (int e = tid; e < 64 * 16; e += 256) {
            int ml = e >> 4, n = e & 15;
            int row = mh * 64 + ml;
            int col = ng * 16 + n;
            float gi = gbuf[0][ml][n];
            float gf = gbuf[1][ml][n];
            float gg = gbuf[2][ml][n];
            float go = gbuf[3][ml][n];
            size_t ci = (size_t)row * HID + col;
            float cn = sigm(gf) * c[ci] + sigm(gi) * fast_tanh(gg);
            float hn = sigm(go) * fast_tanh(cn);
            c[ci]  = cn;
            hw[ci] = (__bf16)hn;
        }

        // ---- prefetch next timestep's bf16 x slab for this block's rows ----
        if (t + 1 < SEQ) {
            for (int p = tid; p < 512; p += 256) {
                int ml = p >> 3, seg = p & 7;               // 8 x 128B lines per row
                const __bf16* addr =
                    xb + ((size_t)(mh * 64 + ml) * SEQ + (t + 1)) * INF + seg * 64;
                __builtin_prefetch(addr, 0, 1);             // global_prefetch_b8
            }
        }

        __threadfence();
        __syncthreads();

        // ---- grid-wide sense barrier (bar[0]=count, bar[1]=generation) ----
        if (tid == 0) {
            int gen = __hip_atomic_load(&bar[1], __ATOMIC_ACQUIRE,
                                        __HIP_MEMORY_SCOPE_AGENT);
            if (__hip_atomic_fetch_add(&bar[0], 1, __ATOMIC_ACQ_REL,
                                       __HIP_MEMORY_SCOPE_AGENT) == NBLK - 1) {
                __hip_atomic_store(&bar[0], 0, __ATOMIC_RELAXED,
                                   __HIP_MEMORY_SCOPE_AGENT);
                __hip_atomic_fetch_add(&bar[1], 1, __ATOMIC_RELEASE,
                                       __HIP_MEMORY_SCOPE_AGENT);
            } else {
                while (__hip_atomic_load(&bar[1], __ATOMIC_ACQUIRE,
                                         __HIP_MEMORY_SCOPE_AGENT) == gen) {
                    __builtin_amdgcn_s_sleep(2);
                }
            }
        }
        __syncthreads();
    }

    // ---- final Dense(1): out[b] = h_T[b,:] . Wd + bd  (h_T lives in hA) ----
    if (blockIdx.x == 0 && tid < BB) {
        const __bf16* hT = hA + (size_t)tid * HID;
        float s = 0.f;
        #pragma unroll 8
        for (int k = 0; k < HID; ++k) s += (float)hT[k] * Wd[k];
        out[tid] = s + bd[0];
    }
}

// ---------------------------------------------------------------------------
// Host launcher
// ---------------------------------------------------------------------------
extern "C" void kernel_launch(void* const* d_in, const int* in_sizes, int n_in,
                              void* d_out, int out_size, void* d_ws, size_t ws_size,
                              hipStream_t stream) {
    const float* x  = (const float*)d_in[0];
    const float* Wi = (const float*)d_in[1];
    const float* Wh = (const float*)d_in[2];
    const float* b  = (const float*)d_in[3];
    const float* Wd = (const float*)d_in[4];
    const float* bd = (const float*)d_in[5];
    float* out = (float*)d_out;

    char* ws = (char*)d_ws;
    __bf16* Wp  = (__bf16*)ws;                                 // 4 MB packed W
    __bf16* xbf = (__bf16*)(ws + (size_t)WP_ELEMS * 2);        // 128 MB bf16 x
    __bf16* hA  = (__bf16*)((char*)xbf + (size_t)X_ELEMS * 2); // 128 KB
    __bf16* hB  = hA + BB * HID;                               // 128 KB
    float*  cS  = (float*)(hB + BB * HID);                     // 256 KB
    int*    bar = (int*)(cS + BB * HID);

    pack_weights_kernel<<<WP_ELEMS / 256, 256, 0, stream>>>(Wi, Wh, Wp);
    convert_x_kernel<<<X_ELEMS / (256 * 4), 256, 0, stream>>>(x, xbf);
    init_state_kernel<<<(BB * HID + 255) / 256, 256, 0, stream>>>(
        (unsigned*)hA, cS, bar);

    size_t smem = LDS_B_BYTES + LDS_G_BYTES;   // 144 KB (<< 320 KB per WGP)
    hipFuncSetAttribute((const void*)lstm_persistent_kernel,
                        hipFuncAttributeMaxDynamicSharedMemorySize, (int)smem);
    lstm_persistent_kernel<<<NBLK, 256, smem, stream>>>(
        xbf, b, Wd, bd, Wp, hA, hB, cS, out, bar);
}
